// MultiheadAttention_59940563583324
// MI455X (gfx1250) — matrix operand order repaired and checked by reading, not compiled, not run on hardware
//
#include <hip/hip_runtime.h>
#include <hip/hip_bf16.h>

#define DEVINL __device__ __forceinline__

typedef __attribute__((ext_vector_type(16))) __bf16 v16bf;
typedef __attribute__((ext_vector_type(4)))  __bf16 v4bf;
typedef __attribute__((ext_vector_type(8)))  float  v8f;

constexpr int B_  = 2;
constexpr int S_  = 2048;
constexpr int H_  = 16;
constexpr int HD_ = 64;
constexpr int D_  = 1024;     // H_ * HD_
constexpr int M_  = B_ * S_;  // 4096

DEVINL v8f wmma_bf16f32(v16bf a, v16bf b, v8f c) {
  return __builtin_amdgcn_wmma_f32_16x16x32_bf16(
      false, a, false, b, (short)0, c, false, false);
}

// Async copy of 16 bytes global -> LDS (ASYNCcnt-tracked, no VGPR data path).
DEVINL void async_b128(unsigned lds_addr, const __bf16* gsrc) {
  asm volatile("global_load_async_to_lds_b128 %0, %1, off"
               :: "v"(lds_addr), "v"(gsrc)
               : "memory");
}
DEVINL void wait_async0() {
  asm volatile("s_wait_asynccnt 0x0" ::: "memory");
}
DEVINL unsigned lds_off(const void* p) {  // low 32 bits of generic = LDS addr
  return (unsigned)(uintptr_t)p;
}

// A fragment (16x32 bf16), tile row-major [16][>=32], row = lane%16.
// VGPR j holds K = 2*(j&3) + (j>=4 ? 16 : 0) + (lane>=16 ? 8 : 0), and K+1.
DEVINL v16bf load_a_frag(const __bf16* tile, int ld, int lane) {
  v16bf f;
  const __bf16* p = tile + (lane & 15) * ld;
  const int koff = (lane >> 4) << 3;
#pragma unroll
  for (int j = 0; j < 8; ++j) {
    const int k = ((j & 3) << 1) + ((j >> 2) << 4) + koff;
    f[2 * j]     = p[k];
    f[2 * j + 1] = p[k + 1];
  }
  return f;
}

// B fragment (32x16 bf16), tile stored K-major per column: tile[col][k].
// col = lane%16; lane holds K = (lane>=16?8:0) + {0..7, 16..23} (table order).
DEVINL v16bf load_b_frag(const __bf16* tile, int ld, int lane) {
  v16bf f;
  const __bf16* p = tile + (lane & 15) * ld + ((lane >> 4) << 3);
#pragma unroll
  for (int j = 0; j < 16; ++j) f[j] = p[j < 8 ? j : j + 8];
  return f;
}

// ---------------------------------------------------------------------------
// Pre-pass: fp32 -> bf16 elementwise (for the x input)
// ---------------------------------------------------------------------------
__global__ __launch_bounds__(256) void cvt_bf16_k(const float* __restrict__ src,
                                                  __bf16* __restrict__ dst) {
  const int i = blockIdx.x * 256 + threadIdx.x;  // one float4 per thread
  const float4 v = ((const float4*)src)[i];
  v4bf o = {(__bf16)v.x, (__bf16)v.y, (__bf16)v.z, (__bf16)v.w};
  ((v4bf*)dst)[i] = o;
}

// Pre-pass: W[K][N] fp32 -> Wt[N][K] bf16 (32x32 LDS tiles)
__global__ __launch_bounds__(256) void transpose_cvt_k(const float* __restrict__ W,
                                                       __bf16* __restrict__ Wt) {
  __shared__ __bf16 t[32][33];
  const int k0 = blockIdx.y * 32, n0 = blockIdx.x * 32;
  const int tid = threadIdx.x;
  const int r = tid >> 3, c = (tid & 7) << 2;
  const float4 v = *(const float4*)(W + (size_t)(k0 + r) * D_ + n0 + c);
  t[c + 0][r] = (__bf16)v.x;
  t[c + 1][r] = (__bf16)v.y;
  t[c + 2][r] = (__bf16)v.z;
  t[c + 3][r] = (__bf16)v.w;
  __syncthreads();
  v4bf o = {t[r][c], t[r][c + 1], t[r][c + 2], t[r][c + 3]};
  *(v4bf*)(Wt + (size_t)(n0 + r) * D_ + k0 + c) = o;
}

// ---------------------------------------------------------------------------
// GEMM: C[M_ x D_] = A[M_ x D_](bf16) @ Wt[D_ x D_]^T(bf16, [n][k]) + bias
// Double-buffered LDS, async global->LDS staging, 8 WMMAs / wave / k-step.
//   OUT_QKV=true : write bf16 [B,H,S,hd];  false: write fp32 [M_, D_]
// ---------------------------------------------------------------------------
template <bool OUT_QKV>
__global__ __launch_bounds__(256) void gemm_wmma_k(
    const __bf16* __restrict__ A, const __bf16* __restrict__ Wt,
    const float* __restrict__ bias,
    __bf16* __restrict__ out_qkv, float* __restrict__ out_f) {
  constexpr int BM = 128, BN = 128, BK = 32, LD = BK + 8;  // 80B pitch (16B mult)
  __shared__ alignas(16) __bf16 As[2][BM][LD];
  __shared__ alignas(16) __bf16 Bs[2][BN][LD];

  const int tid = threadIdx.x, lane = tid & 31, wave = tid >> 5;
  const int wm = wave & 3;   // 4 waves along M -> 32 rows each
  const int wn = wave >> 2;  // 2 waves along N -> 64 cols each
  const int m0 = blockIdx.y * BM, n0 = blockIdx.x * BN;

  // Stage one 128x32 A tile + 128x32 B tile: 4 async B128 copies per thread.
  auto stage = [&](int k0, int buf) {
    const unsigned abase = lds_off(&As[buf][0][0]);
    const unsigned bbase = lds_off(&Bs[buf][0][0]);
#pragma unroll
    for (int i = 0; i < 2; ++i) {
      const int c = tid + 256 * i;           // chunk 0..511 (16B each)
      const int row = c >> 2, q = c & 3;     // 4 chunks per 32-elem row
      const unsigned loff = (unsigned)(row * LD + q * 8) * 2;
      async_b128(abase + loff, A  + (size_t)(m0 + row) * D_ + k0 + q * 8);
      async_b128(bbase + loff, Wt + (size_t)(n0 + row) * D_ + k0 + q * 8);
    }
  };

  const v8f z8 = {0.f, 0.f, 0.f, 0.f, 0.f, 0.f, 0.f, 0.f};
  v8f acc[2][4];
#pragma unroll
  for (int i = 0; i < 2; ++i)
#pragma unroll
    for (int j = 0; j < 4; ++j) acc[i][j] = z8;

  stage(0, 0);
  for (int k0 = 0, it = 0; k0 < D_; k0 += BK, ++it) {
    const int buf = it & 1;
    wait_async0();        // current buffer's async copies complete (this wave)
    __syncthreads();      // ... and everyone else's
    if (k0 + BK < D_) stage(k0 + BK, buf ^ 1);  // prefetch next, overlapped

    v16bf a[2], bfr[4];
#pragma unroll
    for (int mi = 0; mi < 2; ++mi)
      a[mi] = load_a_frag(&As[buf][32 * wm + 16 * mi][0], LD, lane);
#pragma unroll
    for (int ni = 0; ni < 4; ++ni)
      bfr[ni] = load_b_frag(&Bs[buf][64 * wn + 16 * ni][0], LD, lane);
#pragma unroll
    for (int mi = 0; mi < 2; ++mi)
#pragma unroll
      for (int ni = 0; ni < 4; ++ni)
        acc[mi][ni] = wmma_bf16f32(a[mi], bfr[ni], acc[mi][ni]);
  }

  // ---- epilogue: bias + store (C/D layout: row = r + (lane>=16?8:0)) ----
#pragma unroll
  for (int mi = 0; mi < 2; ++mi) {
#pragma unroll
    for (int ni = 0; ni < 4; ++ni) {
      const int col = n0 + 64 * wn + 16 * ni + (lane & 15);
      const float bv = bias[col];
      const int rbase = m0 + 32 * wm + 16 * mi + ((lane >> 4) << 3);
#pragma unroll
      for (int r = 0; r < 8; ++r) {
        const int m = rbase + r;
        const float val = acc[mi][ni][r] + bv;
        if constexpr (OUT_QKV) {
          const int b = m >> 11, s = m & (S_ - 1);
          const int h = col >> 6, d = col & (HD_ - 1);
          out_qkv[((size_t)(b * H_ + h) * S_ + s) * HD_ + d] = (__bf16)val;
        } else {
          out_f[(size_t)m * D_ + col] = val;
        }
      }
    }
  }
}

// ---------------------------------------------------------------------------
// Flash attention: one WG per (qblock=128 rows, head, batch); 8 waves,
// each wave owns 16 query rows. Online softmax, causal.
// Q/K/V bf16 in [B,H,S,hd]; O bf16 in [B,S,H*hd].
// ---------------------------------------------------------------------------
__global__ __launch_bounds__(256) void attn_k(const __bf16* __restrict__ Q,
                                              const __bf16* __restrict__ Kk,
                                              const __bf16* __restrict__ V,
                                              __bf16* __restrict__ O) {
  constexpr int NK = 32, LDK = HD_ + 8, LDV = NK + 8, LDP = NK + 8;
  __shared__ alignas(16) __bf16 Ks[NK][LDK];    // [key][d]  (async-staged)
  __shared__ alignas(16) __bf16 VT[HD_][LDV];   // [d][key]  (transposed V)
  __shared__ alignas(16) __bf16 Ps[8][16][LDP]; // per-wave P (C->A relayout)

  const int tid = threadIdx.x, lane = tid & 31, wave = tid >> 5;
  const int qb = blockIdx.x, h = blockIdx.y, b = blockIdx.z;
  const int qbase = qb * 128 + wave * 16;
  const size_t headoff = (size_t)(b * H_ + h) * S_ * HD_;
  const __bf16* qh = Q + headoff;
  const __bf16* kh = Kk + headoff;
  const __bf16* vh = V + headoff;
  const unsigned ksbase = lds_off(&Ks[0][0]);

  // Q fragments for hd=64 (two 16x32 A-fragments), kept in registers
  v16bf qf[2];
  {
    const __bf16* p = qh + (size_t)(qbase + (lane & 15)) * HD_;
    const int koff = (lane >> 4) << 3;
#pragma unroll
    for (int ks = 0; ks < 2; ++ks)
#pragma unroll
      for (int j = 0; j < 8; ++j) {
        const int d = 32 * ks + ((j & 3) << 1) + ((j >> 2) << 4) + koff;
        qf[ks][2 * j]     = p[d];
        qf[ks][2 * j + 1] = p[d + 1];
      }
  }

  const v8f z8 = {0.f, 0.f, 0.f, 0.f, 0.f, 0.f, 0.f, 0.f};
  v8f acc[4];
#pragma unroll
  for (int ni = 0; ni < 4; ++ni) acc[ni] = z8;
  float mrow[8], lrow[8];
#pragma unroll
  for (int r = 0; r < 8; ++r) { mrow[r] = -1e30f; lrow[r] = 0.f; }

  const int nkb = (qb + 1) * (128 / NK);  // causal: keys < (qb+1)*128
  for (int kb = 0; kb < nkb; ++kb) {
    const int kstart = kb * NK;
    __syncthreads();  // previous tiles fully consumed
    {
      const int key = tid >> 3, d0 = (tid & 7) << 3;  // 8 bf16 = 16B chunk
      // K tile: straight copy -> async global->LDS
      async_b128(ksbase + (unsigned)(key * LDK + d0) * 2,
                 kh + (size_t)(kstart + key) * HD_ + d0);
      // V tile: transpose through registers
      const __bf16* vsrc = vh + (size_t)(kstart + key) * HD_ + d0;
#pragma unroll
      for (int i = 0; i < 8; ++i) VT[d0 + i][key] = vsrc[i];
    }
    wait_async0();
    __syncthreads();
    if (kstart > qbase + 15) continue;  // whole block above the diagonal

    // ---- scores: S = Q @ K^T (two 16x16 key tiles, two k-steps each) ----
    v8f sc[2];
#pragma unroll
    for (int t = 0; t < 2; ++t) {
      sc[t] = z8;
#pragma unroll
      for (int ks = 0; ks < 2; ++ks) {
        v16bf kf = load_b_frag(&Ks[16 * t][32 * ks], LDK, lane);
        sc[t] = wmma_bf16f32(qf[ks], kf, sc[t]);
      }
    }

    // ---- scale + causal mask + online softmax (rows across 16 lanes) ----
    float alpha[8];
#pragma unroll
    for (int r = 0; r < 8; ++r) {
      const int qidx = qbase + r + ((lane >> 4) << 3);
#pragma unroll
      for (int t = 0; t < 2; ++t) {
        const int kidx = kstart + 16 * t + (lane & 15);
        const float v = sc[t][r] * 0.125f;  // 1/sqrt(64)
        sc[t][r] = (kidx <= qidx) ? v : -1e30f;
      }
      float mx = fmaxf(sc[0][r], sc[1][r]);
#pragma unroll
      for (int off = 1; off < 16; off <<= 1)
        mx = fmaxf(mx, __shfl_xor(mx, off, 16));
      const float mnew = fmaxf(mrow[r], mx);
      alpha[r] = __expf(mrow[r] - mnew);
      mrow[r] = mnew;
      const float p0 = __expf(sc[0][r] - mnew);
      const float p1 = __expf(sc[1][r] - mnew);
      sc[0][r] = p0;
      sc[1][r] = p1;
      float sum = p0 + p1;
#pragma unroll
      for (int off = 1; off < 16; off <<= 1)
        sum += __shfl_xor(sum, off, 16);
      lrow[r] = lrow[r] * alpha[r] + sum;
    }
#pragma unroll
    for (int ni = 0; ni < 4; ++ni)
#pragma unroll
      for (int r = 0; r < 8; ++r) acc[ni][r] *= alpha[r];

    // ---- restage P (C layout -> A layout) through per-wave LDS ----
#pragma unroll
    for (int r = 0; r < 8; ++r) {
      const int prow = r + ((lane >> 4) << 3);
      Ps[wave][prow][(lane & 15)]      = (__bf16)sc[0][r];
      Ps[wave][prow][16 + (lane & 15)] = (__bf16)sc[1][r];
    }
    asm volatile("s_wait_dscnt 0" ::: "memory");  // wave-private RAW fence
    v16bf pf = load_a_frag(&Ps[wave][0][0], LDP, lane);

    // ---- O += P @ V ----
#pragma unroll
    for (int ni = 0; ni < 4; ++ni) {
      v16bf vf = load_b_frag(&VT[16 * ni][0], LDV, lane);
      acc[ni] = wmma_bf16f32(pf, vf, acc[ni]);
    }
  }

  // ---- epilogue: normalize, store to [B,S,H*hd] for the output GEMM ----
#pragma unroll
  for (int ni = 0; ni < 4; ++ni) {
    const int d = 16 * ni + (lane & 15);
#pragma unroll
    for (int r = 0; r < 8; ++r) {
      const int s = qbase + r + ((lane >> 4) << 3);
      const float val = acc[ni][r] / lrow[r];
      O[((size_t)(b * S_ + s) * H_ + h) * HD_ + d] = (__bf16)val;
    }
  }
}

// ---------------------------------------------------------------------------
extern "C" void kernel_launch(void* const* d_in, const int* in_sizes, int n_in,
                              void* d_out, int out_size, void* d_ws,
                              size_t ws_size, hipStream_t stream) {
  (void)in_sizes; (void)n_in; (void)out_size; (void)ws_size;
  const float* x  = (const float*)d_in[0];
  const float* wq = (const float*)d_in[1];
  const float* bq = (const float*)d_in[2];
  const float* wk = (const float*)d_in[3];
  const float* bk = (const float*)d_in[4];
  const float* wv = (const float*)d_in[5];
  const float* bv = (const float*)d_in[6];
  const float* wo = (const float*)d_in[7];
  const float* bo = (const float*)d_in[8];
  float* out = (float*)d_out;

  __bf16* ws = (__bf16*)d_ws;
  const size_t per = (size_t)M_ * D_;   // 4M elems
  const size_t wsz = (size_t)D_ * D_;   // 1M elems
  __bf16* Qw  = ws;
  __bf16* Kw  = Qw + per;
  __bf16* Vw  = Kw + per;
  __bf16* Ow  = Vw + per;
  __bf16* Xb  = Ow + per;
  __bf16* Wqt = Xb + per;
  __bf16* Wkt = Wqt + wsz;
  __bf16* Wvt = Wkt + wsz;
  __bf16* Wot = Wvt + wsz;

  const dim3 tb(256, 1, 1);
  const dim3 gw(D_ / 32, D_ / 32, 1);  // weight transpose tiles

  // Pre-pass: bf16 conversions + weight transposes (one-shot, L2-resident)
  cvt_bf16_k<<<dim3((M_ * D_) / 1024), tb, 0, stream>>>(x, Xb);
  transpose_cvt_k<<<gw, tb, 0, stream>>>(wq, Wqt);
  transpose_cvt_k<<<gw, tb, 0, stream>>>(wk, Wkt);
  transpose_cvt_k<<<gw, tb, 0, stream>>>(wv, Wvt);
  transpose_cvt_k<<<gw, tb, 0, stream>>>(wo, Wot);

  const dim3 gb(D_ / 128, M_ / 128, 1);  // (8, 32)
  gemm_wmma_k<true><<<gb, tb, 0, stream>>>(Xb, Wqt, bq, Qw, nullptr);
  gemm_wmma_k<true><<<gb, tb, 0, stream>>>(Xb, Wkt, bk, Kw, nullptr);
  gemm_wmma_k<true><<<gb, tb, 0, stream>>>(Xb, Wvt, bv, Vw, nullptr);

  attn_k<<<dim3(S_ / 128, H_, B_), tb, 0, stream>>>(Qw, Kw, Vw, Ow);

  gemm_wmma_k<false><<<gb, tb, 0, stream>>>(Ow, Wot, bo, nullptr, out);
}
